// ActorsHead_52561809768759
// MI455X (gfx1250) — matrix operand order repaired and checked by reading, not compiled, not run on hardware
//
#include <hip/hip_runtime.h>
#include <hip/hip_bf16.h>
#include <math.h>

// ---------------------------------------------------------------------------
// Types
// ---------------------------------------------------------------------------
typedef __attribute__((ext_vector_type(16))) __bf16 v16bf;
typedef __attribute__((ext_vector_type(8)))  float  v8f;
typedef __attribute__((ext_vector_type(4)))  float  f4;

#define NE       8192
#define LN_EPS   1e-5f

struct U2 { unsigned x, y; };

__device__ __forceinline__ unsigned rotl32(unsigned v, int r) {
  return (v << r) | (v >> (32 - r));
}

// Exact JAX threefry2x32 (5 groups of 4 rounds, ks2 = 0x1BD11BDA ^ k0 ^ k1)
__device__ __forceinline__ U2 threefry2x32(unsigned k0, unsigned k1,
                                           unsigned x0, unsigned x1) {
  unsigned k2 = 0x1BD11BDAu ^ k0 ^ k1;
  x0 += k0; x1 += k1;
#define TF_R4(a,b,c,d) \
  x0 += x1; x1 = rotl32(x1,(a)); x1 ^= x0; \
  x0 += x1; x1 = rotl32(x1,(b)); x1 ^= x0; \
  x0 += x1; x1 = rotl32(x1,(c)); x1 ^= x0; \
  x0 += x1; x1 = rotl32(x1,(d)); x1 ^= x0;
  TF_R4(13,15,26, 6)  x0 += k1; x1 += k2 + 1u;
  TF_R4(17,29,16,24)  x0 += k2; x1 += k0 + 2u;
  TF_R4(13,15,26, 6)  x0 += k0; x1 += k1 + 3u;
  TF_R4(17,29,16,24)  x0 += k1; x1 += k2 + 4u;
  TF_R4(13,15,26, 6)  x0 += k2; x1 += k0 + 5u;
#undef TF_R4
  U2 r; r.x = x0; r.y = x1; return r;
}

// wave32 reductions
__device__ __forceinline__ float wave_sum32(float v) {
  v += __shfl_xor(v, 16, 32);
  v += __shfl_xor(v,  8, 32);
  v += __shfl_xor(v,  4, 32);
  v += __shfl_xor(v,  2, 32);
  v += __shfl_xor(v,  1, 32);
  return v;
}

__device__ __forceinline__ float lnorm32(float v, float g, float b) {
  float m   = wave_sum32(v) * (1.f / 32.f);
  float d   = v - m;
  float var = wave_sum32(d * d) * (1.f / 32.f);
  return d * (1.f / sqrtf(var + LN_EPS)) * g + b;
}

__device__ __forceinline__ float sigmoidf(float x) { return 1.f / (1.f + expf(-x)); }

// Wave-relative LDS byte offset: per CDNA5 aperture decode, the low 32 bits of
// a generic pointer into LDS are the LDS address.
__device__ __forceinline__ unsigned lds_off_of(const void* p) {
  return (unsigned)(uintptr_t)p;
}

// ---------------------------------------------------------------------------
// Kernel 1: streaming zero-fill of the whole output (268 MB). NT stores so we
// do not thrash the 192MB L2 that must keep entity_keys/W0/W3 hot.
// ---------------------------------------------------------------------------
__global__ __launch_bounds__(256) void ah_zero_fill(float* __restrict__ out, long n) {
  long n4 = n >> 2;
  f4* p = (f4*)out;
  f4 z = {0.f, 0.f, 0.f, 0.f};
  long stride = (long)gridDim.x * 256;
  for (long i = (long)blockIdx.x * 256 + threadIdx.x; i < n4; i += stride)
    __builtin_nontemporal_store(z, p + i);
  // tail (out_size is 4-divisible in practice, keep it robust)
  long tail = n & 3;
  if (blockIdx.x == 0 && threadIdx.x < tail) out[n - 1 - threadIdx.x] = 0.f;
}

// ---------------------------------------------------------------------------
// Kernel 2: entity_keys = enc(8192x256) @ Wk^T(256x32) + bk  via split-bf16
// WMMA (hi/lo decomposition, 3x v_wmma_f32_16x16x32_bf16 per K-step).
// A tiles are staged into LDS with the CDNA5 async-to-LDS engine
// (GLOBAL_LOAD_ASYNC_TO_LDS_B32, ASYNCcnt-tracked, no VGPR round trip),
// then gathered into the documented 16-bit A fragment layout.
// Output layout: keys[m][n], m in [0,8192), n in [0,32)  (row per entity).
// ---------------------------------------------------------------------------
__global__ __launch_bounds__(64) void ah_keys_wmma(const float* __restrict__ enc,
                                                   const float* __restrict__ Wk,
                                                   const float* __restrict__ bk,
                                                   float* __restrict__ keys) {
  __shared__ float As[16][33];          // +1 pad vs bank conflicts
  const int t     = threadIdx.x;
  const int lane  = t & 31;
  const int ntile = t >> 5;             // 0 or 1
  const int mtile = blockIdx.x;         // 0..511
  const int hi16  = (lane < 16) ? 0 : 1;
  const int abase = hi16 ? 8  : 0;      // A K-group base for this half-wave
  const int bbase = hi16 ? 8 : 0;       // B K base for this half-wave
  const int arow  = lane & 15;          // A row within tile (M = 0..15)
  const int ncol  = ntile * 16 + (lane & 15);  // output column 0..31

  const unsigned lds_base = lds_off_of(&As[0][0]);

  v8f acc = {};
  for (int kk = 0; kk < 256; kk += 32) {
    // stage A 16x32 fp32 tile via async global->LDS DMA (8 dwords/lane)
    for (int j = 0; j < 8; ++j) {
      int idx = j * 64 + t;             // 0..511
      int r = idx >> 5, c = idx & 31;
      unsigned ldsoff = lds_base + (unsigned)((r * 33 + c) * 4);
      unsigned goff   = (unsigned)((((mtile * 16 + r) * 256) + kk + c) * 4);
      asm volatile("global_load_async_to_lds_b32 %0, %1, %2"
                   :: "v"(ldsoff), "v"(goff), "s"(enc)
                   : "memory");
    }
    // each wave drains its own ASYNCcnt, then the barrier publishes the tile
    asm volatile("s_wait_asynccnt 0" ::: "memory");
    __syncthreads();

    v16bf ah, al, bh, bl;
    for (int i = 0; i < 16; ++i) {
      // A layout: lanes 0-15 K={0..7,16..23}, lanes 16-31 K={8..15,24..31}
      int klocal = abase + i + ((i < 8) ? 0 : 8);
      float x  = As[arow][klocal];
      __bf16 h = (__bf16)x;
      ah[i] = h;
      al[i] = (__bf16)(x - (float)h);
      // B layout: lane n<16 -> col n, K={0..7,16..23} ; lane>=16 -> col n-16, K={8..15,24..31}
      // B[k][n] = Wk[n][k]
      float y   = Wk[(size_t)ncol * 256 + kk + bbase + i + ((i < 8) ? 0 : 8)];
      __bf16 hy = (__bf16)y;
      bh[i] = hy;
      bl[i] = (__bf16)(y - (float)hy);
    }
    // fp32-class precision: hi*hi + hi*lo + lo*hi
    acc = __builtin_amdgcn_wmma_f32_16x16x32_bf16(false, ah, false, bh, (short)0, acc, false, false);
    acc = __builtin_amdgcn_wmma_f32_16x16x32_bf16(false, ah, false, bl, (short)0, acc, false, false);
    acc = __builtin_amdgcn_wmma_f32_16x16x32_bf16(false, al, false, bh, (short)0, acc, false, false);
    __syncthreads();
  }

  // C/D layout: lane L, VGPR r -> M = r + (L<16 ? 0 : 8), N = L&15
  float bias = bk[ncol];
  int m0 = mtile * 16 + (hi16 ? 8 : 0);
  for (int r = 0; r < 8; ++r)
    keys[(size_t)(m0 + r) * 32 + ncol] = acc[r] + bias;
}

// ---------------------------------------------------------------------------
// Kernel 3: the 64 sequential autoregressive steps. One persistent workgroup
// of 1024 threads (32 wave32s on one WGP). All carried state lives in LDS;
// entity/selection masks are bit-packed. keys (1MB) stream from L2 per step.
// ---------------------------------------------------------------------------
__global__ __launch_bounds__(1024) void ah_step_loop(
    const float* __restrict__ utype,   // 233
    const float* __restrict__ emask_in,// 8192
    const float* __restrict__ arenc,   // 1024
    const int*   __restrict__ uct,     // 1
    const float* __restrict__ We,  const float* __restrict__ be,
    const float* __restrict__ W0,  const float* __restrict__ b0,
    const float* __restrict__ W1,  const float* __restrict__ b1,
    const float* __restrict__ Wf,  const float* __restrict__ bfv,
    const float* __restrict__ Wi0, const float* __restrict__ bi0,
    const float* __restrict__ Wi1, const float* __restrict__ bi1,
    const float* __restrict__ Wo,  const float* __restrict__ bo,
    const float* __restrict__ W3,  const float* __restrict__ b3,
    const float* __restrict__ lng, const float* __restrict__ lnb,
    const float* __restrict__ keys,    // (8192,32) in workspace
    float* __restrict__ out)           // [E*E logits | E sel | 1024 ar]
{
  __shared__ float    s_ar[1024];
  __shared__ float    s_fe[256];
  __shared__ float    s_i0[256];
  __shared__ float    s_x[64];
  __shared__ float    s_q[32];
  __shared__ float    s_h[32];
  __shared__ float    s_selc[32];
  __shared__ unsigned s_emask[256];    // bit-packed entity mask
  __shared__ unsigned s_selb[256];     // bit-packed selection
  __shared__ unsigned s_keybits[128];  // threefry split() output words
  __shared__ float    s_redf[32];
  __shared__ int      s_redi[32];
  __shared__ float    s_tot;
  __shared__ int      s_pick;
  __shared__ int      s_done;
  __shared__ int      s_updok;

  const int t   = threadIdx.x;
  const int lid = t & 31;
  const int wid = t >> 5;

  int ns = uct[0];
  ns = ns < 0 ? 0 : (ns > 64 ? 64 : ns);

  // ---------------- init ----------------
  if (t < 256) {
    unsigned w = 0;
    for (int b = 0; b < 32; ++b)
      if (emask_in[t * 32 + b] != 0.f) w |= (1u << b);
    s_emask[t] = w;
    s_selb[t]  = 0u;
    // func_embed = relu(utype @ We + be)
    float acc = be[t];
    for (int k = 0; k < 233; ++k) acc += utype[k] * We[k * 256 + t];
    s_fe[t] = fmaxf(acc, 0.f);
  }
  s_ar[t] = arenc[t];
  if (t < 32) { s_q[t] = 0.f; s_h[t] = 0.f; }
  if (t == 0) s_done = 0;
  // JAX split(key(42), 64): threefry(key, iota(128)), halves layout
  if (t < 64) {
    U2 r = threefry2x32(0u, 42u, (unsigned)t, (unsigned)(64 + t));
    s_keybits[t]      = r.x;
    s_keybits[64 + t] = r.y;
  }
  __syncthreads();

  // ---------------- sequential steps ----------------
  for (int s = 0; s < ns; ++s) {
    int active = !s_done;

    // (a) i0 = ar @ W0 + b0 + func_embed           (256 outputs)
    if (t < 256) {
      float acc = b0[t] + s_fe[t];
      for (int k = 0; k < 1024; ++k) acc += s_ar[k] * W0[k * 256 + t];
      s_i0[t] = acc;
    }
    __syncthreads();

    // (b) i1 = relu(relu(i0) @ W1 + b1); x = [i1, query]   (wave 0)
    if (t < 32) {
      float acc = b1[t];
      for (int k = 0; k < 256; ++k) acc += fmaxf(s_i0[k], 0.f) * W1[k * 32 + t];
      s_x[t]      = fmaxf(acc, 0.f);
      s_x[32 + t] = s_q[t];            // old query
    }
    __syncthreads();

    // (c) LSTM-style gates + layernorms (wave 0, wave32-wide LN)
    if (t < 32) {
      float xf = bfv[t], x0g = bi0[t], x1g = bi1[t], xo = bo[t];
      for (int k = 0; k < 64; ++k) {
        float xv = s_x[k];
        xf  += xv * Wf [k * 32 + t];
        x0g += xv * Wi0[k * 32 + t];
        x1g += xv * Wi1[k * 32 + t];
        xo  += xv * Wo [k * 32 + t];
      }
      float g = lng[t], b = lnb[t];
      float forget = lnorm32(sigmoidf(xf), g, b);
      float rem    = lnorm32(sigmoidf(x0g) * tanhf(x1g), g, b);
      float og     = lnorm32(sigmoidf(xo), g, b);
      float nh = rem + forget * s_h[t];
      float nq = tanhf(nh) * og;
      if (active) { s_h[t] = nh; s_q[t] = nq; }
    }
    __syncthreads();

    // (d) sim = query @ keys^T over 8192 entities; softmax^(1/T); sample
    float vec[8];
    float lsum = 0.f;
    for (int u = 0; u < 8; ++u) {
      int m = u * 1024 + t;
      const float* kr = keys + (size_t)m * 32;
      if (u < 7) __builtin_prefetch(keys + (size_t)(m + 1024) * 32, 0, 1);
      float dot = 0.f;
      for (int j = 0; j < 32; ++j) dot += s_q[j] * kr[j];
      float sg = sigmoidf(dot);
      float v  = (sg > 0.f) ? expf(1.25f * logf(sg)) : 0.f;  // s^(1/0.8)
      if (v != v) v = 0.f;                                    // nan_to_num
      vec[u] = v;
      lsum  += v;
    }
    { // block sum -> s_tot
      float wsv = wave_sum32(lsum);
      if (lid == 0) s_redf[wid] = wsv;
      __syncthreads();
      if (t < 32) {
        float v = wave_sum32(s_redf[t]);
        if (t == 0) s_tot = v;
      }
      __syncthreads();
    }
    float tot   = s_tot;
    int   valid = (tot != 0.f);
    float rinv  = valid ? (1.f / tot) : 0.f;

    unsigned k0 = s_keybits[2 * s], k1 = s_keybits[2 * s + 1];
    float zbest = -INFINITY;
    int   ibest = 0x7fffffff;
    for (int u = 0; u < 8; ++u) {
      int   m    = u * 1024 + t;
      float soft = valid ? vec[u] * rinv : 0.f;
      float row  = (active && valid) ? soft : 0.f;
      __builtin_nontemporal_store(row, out + (size_t)s * NE + m);
      // JAX random_bits(key,(8192,)): halves counter layout
      unsigned c0 = (m < 4096) ? (unsigned)m : (unsigned)(m - 4096);
      U2 rb = threefry2x32(k0, k1, c0, c0 + 4096u);
      unsigned bits = (m < 4096) ? rb.x : rb.y;
      const float tiny = 1.17549435e-38f;
      float uu = __uint_as_float((bits >> 9) | 0x3f800000u) - 1.0f;
      uu = fmaxf(uu * (1.f - tiny) + tiny, tiny);
      float z = logf(soft) + (-logf(-logf(uu)));   // log(soft) + gumbel
      if (z > zbest || (z == zbest && m < ibest)) { zbest = z; ibest = m; }
    }
    { // block argmax (first-max tie-break) -> s_pick
      for (int mk = 16; mk > 0; mk >>= 1) {
        float oz = __shfl_xor(zbest, mk, 32);
        int   oi = __shfl_xor(ibest, mk, 32);
        if (oz > zbest || (oz == zbest && oi < ibest)) { zbest = oz; ibest = oi; }
      }
      if (lid == 0) { s_redf[wid] = zbest; s_redi[wid] = ibest; }
      __syncthreads();
      if (t < 32) {
        float z = s_redf[t]; int i = s_redi[t];
        for (int mk = 16; mk > 0; mk >>= 1) {
          float oz = __shfl_xor(z, mk, 32);
          int   oi = __shfl_xor(i, mk, 32);
          if (oz > z || (oz == z && oi < i)) { z = oz; i = oi; }
        }
        if (t == 0) s_pick = i;
      }
      __syncthreads();
    }

    // (e) selection vector, masks, done flag (wave 0)
    if (t < 32) {
      int pick = s_pick;
      float kv = keys[(size_t)pick * 32 + t];
      float mean = wave_sum32(kv) * (1.f / 32.f);
      float sc = kv - mean;
      s_selc[t] = sc;
      int has_nan = (__ballot(sc != sc) != 0ull);
      if (t == 0) {
        int embit = (s_emask[pick >> 5] >> (pick & 31)) & 1;
        int dosel = active && valid && embit;
        s_updok = dosel && !has_nan;
        if (dosel) {
          s_emask[pick >> 5] &= ~(1u << (pick & 31));
          s_selb [pick >> 5] |=  (1u << (pick & 31));
          if (has_nan) s_done = 1;
        }
      }
    }
    __syncthreads();

    // (f) ar += relu(selection @ W3 + b3)           (1024 outputs)
    if (s_updok) {
      float acc = b3[t];
      for (int k = 0; k < 32; ++k) acc += s_selc[k] * W3[k * 1024 + t];
      s_ar[t] += fmaxf(acc, 0.f);
    }
    __syncthreads();
  }

  // ---------------- outputs: sel (E floats), ar (1024 floats) ----------------
  for (int u = 0; u < 8; ++u) {
    int m = u * 1024 + t;
    float sv = ((s_selb[m >> 5] >> (m & 31)) & 1) ? 1.f : 0.f;
    out[(size_t)NE * NE + m] = sv;
  }
  out[(size_t)NE * NE + NE + t] = s_ar[t];
}

// ---------------------------------------------------------------------------
// Host entry
// ---------------------------------------------------------------------------
extern "C" void kernel_launch(void* const* d_in, const int* in_sizes, int n_in,
                              void* d_out, int out_size, void* d_ws, size_t ws_size,
                              hipStream_t stream) {
  (void)in_sizes; (void)n_in; (void)ws_size;
  const float* utype  = (const float*)d_in[0];
  const float* emask  = (const float*)d_in[1];
  const float* enc    = (const float*)d_in[2];
  const float* arenc  = (const float*)d_in[3];
  const int*   uct    = (const int*)  d_in[4];
  const float* We  = (const float*)d_in[5],  *be  = (const float*)d_in[6];
  const float* Wk  = (const float*)d_in[7],  *bk  = (const float*)d_in[8];
  const float* W0  = (const float*)d_in[9],  *b0  = (const float*)d_in[10];
  const float* W1  = (const float*)d_in[11], *b1  = (const float*)d_in[12];
  const float* Wf  = (const float*)d_in[13], *bfv = (const float*)d_in[14];
  const float* Wi0 = (const float*)d_in[15], *bi0 = (const float*)d_in[16];
  const float* Wi1 = (const float*)d_in[17], *bi1 = (const float*)d_in[18];
  const float* Wo  = (const float*)d_in[19], *bo  = (const float*)d_in[20];
  const float* W3  = (const float*)d_in[21], *b3  = (const float*)d_in[22];
  const float* lng = (const float*)d_in[23], *lnb = (const float*)d_in[24];

  float* out  = (float*)d_out;
  float* keys = (float*)d_ws;          // 8192*32 fp32 = 1 MB scratch

  ah_zero_fill<<<2048, 256, 0, stream>>>(out, (long)out_size);
  ah_keys_wmma<<<512, 64, 0, stream>>>(enc, Wk, bk, keys);
  ah_step_loop<<<1, 1024, 0, stream>>>(utype, emask, arenc, uct,
                                       We, be, W0, b0, W1, b1,
                                       Wf, bfv, Wi0, bi0, Wi1, bi1,
                                       Wo, bo, W3, b3, lng, lnb,
                                       keys, out);
}
